// MultiQueryAttention_37108517437746
// MI455X (gfx1250) — matrix operand order repaired and checked by reading, not compiled, not run on hardware
//
#include <hip/hip_runtime.h>
#include <hip/hip_bf16.h>
#include <stdint.h>

// ---------------------------------------------------------------------------
// MQA forward for MI455X (gfx1250, wave32, WMMA).
// bf16 operands + fp32 accumulation via v_wmma_f32_16x16x32_bf16.
// Tiled GEMM: 8 waves/block, 128x128 tile, K-step 32, double-buffered LDS,
// CDNA5 async global->LDS copies (ASYNCcnt) overlapped with WMMA compute,
// single barrier per K-step.
// ---------------------------------------------------------------------------

typedef __attribute__((ext_vector_type(16))) __bf16 v16bf;
typedef __attribute__((ext_vector_type(8)))  float  v8f;

#define TILE_M 128
#define TILE_N 128
#define TILE_K 32
#define LDS_STRIDE 40   // 32 + 8 pad halfwords: 80B rows -> b128-aligned, even
#define TILE_ELEMS (TILE_M * LDS_STRIDE)

__device__ __forceinline__ unsigned short f32_to_bf16_bits(float f) {
    unsigned int u = __float_as_uint(f);
    u += 0x7FFFu + ((u >> 16) & 1u);   // round-to-nearest-even
    return (unsigned short)(u >> 16);
}

union Frag16 {
    v16bf v;
    unsigned int u[8];
};

// Per-lane async copy of 16B global -> LDS (GLOBAL_LOAD_ASYNC_TO_LDS_B128,
// tracked by ASYNCcnt). LDS byte address = low 32 bits of generic pointer.
__device__ __forceinline__ void async_copy_b128(void* lds_dst, const void* gsrc) {
    unsigned lds_addr = (unsigned)(uintptr_t)lds_dst;
    unsigned long long ga = (unsigned long long)(uintptr_t)gsrc;
    asm volatile("global_load_async_to_lds_b128 %0, %1, off"
                 :: "v"(lds_addr), "v"(ga)
                 : "memory");
}
__device__ __forceinline__ void wait_async_zero() {
    asm volatile("s_wait_asynccnt 0x0" ::: "memory");
}

// C[M,N] = alpha * A[M,K] x B (+bias). TRANS_B: B given as [N,K] row-major
// (Q*K^T). A_F32: A is fp32 in global, converted to bf16 while staging (P*V).
// Batched over blockIdx.z with outer/inner batch stride pairs.
template<bool TRANS_B, bool A_F32, bool OUT_BF16>
__global__ void __launch_bounds__(256)
gemm_wmma_kernel(const void* __restrict__ Ag_,
                 const unsigned short* __restrict__ Bg,
                 const float* __restrict__ bias,
                 void* __restrict__ Cg_,
                 int N, int K, int lda, int ldb, int ldc, float alpha,
                 int innerCount,
                 long long aOuter, long long aInner,
                 long long bOuter, long long bInner,
                 long long cOuter, long long cInner)
{
    constexpr bool A_ASYNC = !A_F32;
    constexpr bool B_ASYNC = TRANS_B;
    constexpr bool ANY_ASYNC = A_ASYNC || B_ASYNC;

    __shared__ __align__(16) unsigned short As[2][TILE_ELEMS];
    __shared__ __align__(16) unsigned short Bs[2][TILE_ELEMS];

    const int tid  = threadIdx.x;
    const int lane = tid & 31;
    const int wave = tid >> 5;
    const int wr   = wave & 3;    // 4 wave rows * 32 M each
    const int wc   = wave >> 2;   // 2 wave cols * 64 N each
    const int lm   = lane & 15;
    const int half = lane >> 4;

    const int m0 = blockIdx.y * TILE_M;
    const int n0 = blockIdx.x * TILE_N;
    const int z  = blockIdx.z;
    const int zo = z / innerCount;
    const int zi = z % innerCount;

    // --- per-thread staging base pointers, advanced by fixed strides ---
    // A bf16 async: chunk i of 2 copies 16B; row = i*64 + tid/4, col16B = tid%4
    const unsigned short* aB =
        (const unsigned short*)Ag_ + zo * aOuter + zi * aInner
        + (long long)(m0 + (tid >> 2)) * lda + (tid & 3) * 8;
    // A fp32 manual: chunk i of 4; row = i*32 + tid/8, col float4 = tid%8
    const float* aF =
        (const float*)Ag_ + zo * aOuter + zi * aInner
        + (long long)(m0 + (tid >> 3)) * lda + (tid & 7) * 4;
    // B trans async: same shape as A bf16 but over n
    const unsigned short* bT =
        Bg + zo * bOuter + zi * bInner
        + (long long)(n0 + (tid >> 2)) * ldb + (tid & 3) * 8;
    // B normal manual: pair col = tid%64, k row = i*4 + tid/64
    const unsigned short* bN =
        Bg + zo * bOuter + zi * bInner
        + (long long)(tid >> 6) * ldb + n0 + (tid & 63) * 2;

    // LDS destination offsets (u16 indices), constant per thread
    const int aDst  = (tid >> 2) * LDS_STRIDE + (tid & 3) * 8;           // async A / B
    const int aDstF = (tid >> 3) * LDS_STRIDE + (tid & 7) * 4;           // fp32 A
    const int bDstN = (tid & 63) * 2 * LDS_STRIDE + (tid >> 6);          // manual B

    float4   avReg[4];   // in-flight fp32 A tile (held across compute)
    unsigned bReg[8];    // in-flight manual B tile

    // ---- stage phase 1: issue global traffic for one tile ----
    auto issueA = [&](int nb) {
        if (A_F32) {
#pragma unroll
            for (int i = 0; i < 4; ++i)
                avReg[i] = *(const float4*)(aF + i * (32LL * lda));
            aF += TILE_K;
        } else {
            async_copy_b128(&As[nb][aDst], aB);
            async_copy_b128(&As[nb][aDst + 64 * LDS_STRIDE], aB + 64LL * lda);
            aB += TILE_K;
        }
    };
    auto issueB = [&](int nb) {
        if (TRANS_B) {
            async_copy_b128(&Bs[nb][aDst], bT);
            async_copy_b128(&Bs[nb][aDst + 64 * LDS_STRIDE], bT + 64LL * ldb);
            bT += TILE_K;
        } else {
#pragma unroll
            for (int i = 0; i < 8; ++i)
                bReg[i] = *(const unsigned int*)(bN + i * (4LL * ldb));
            bN += 32LL * ldb;
        }
    };
    // ---- stage phase 2: commit register-staged data to LDS ----
    auto commitA = [&](int nb) {
        if (A_F32) {
#pragma unroll
            for (int i = 0; i < 4; ++i) {
                unsigned p0 = (unsigned)f32_to_bf16_bits(avReg[i].x) |
                              ((unsigned)f32_to_bf16_bits(avReg[i].y) << 16);
                unsigned p1 = (unsigned)f32_to_bf16_bits(avReg[i].z) |
                              ((unsigned)f32_to_bf16_bits(avReg[i].w) << 16);
                uint2 w; w.x = p0; w.y = p1;
                *(uint2*)&As[nb][aDstF + i * 32 * LDS_STRIDE] = w;
            }
        }
    };
    auto commitB = [&](int nb) {
        if (!TRANS_B) {
#pragma unroll
            for (int i = 0; i < 8; ++i) {
                Bs[nb][bDstN + i * 4]              = (unsigned short)(bReg[i] & 0xffffu);
                Bs[nb][bDstN + LDS_STRIDE + i * 4] = (unsigned short)(bReg[i] >> 16);
            }
        }
    };

    const v8f vzero = {0.f, 0.f, 0.f, 0.f, 0.f, 0.f, 0.f, 0.f};
    v8f acc[2][4];
#pragma unroll
    for (int tm = 0; tm < 2; ++tm)
#pragma unroll
        for (int tn = 0; tn < 4; ++tn)
            acc[tm][tn] = vzero;

    const int nk = K / TILE_K;

    // prologue: fill buffer 0
    issueA(0); issueB(0);
    commitA(0); commitB(0);
    if (ANY_ASYNC) wait_async_zero();
    __syncthreads();

    for (int it = 0; it < nk; ++it) {
        const int cur = it & 1, nxt = cur ^ 1;
        const bool hasNext = (it + 1 < nk);

        // issue next tile's traffic (async engine + global loads run
        // concurrently with this tile's WMMAs)
        if (hasNext) { issueA(nxt); issueB(nxt); }

        // ---- fragments per ISA 7.12.2 layouts, 8 WMMAs per K-step ----
        Frag16 bfrag[4];
#pragma unroll
        for (int tn = 0; tn < 4; ++tn) {
            const int nbase = (wc * 64 + tn * 16 + lm) * LDS_STRIDE;
#pragma unroll
            for (int v = 0; v < 8; ++v)   // B 32x16: k = (v/4)*16 + half*8 + (v%4)*2
                bfrag[tn].u[v] = *(const unsigned int*)&Bs[cur][nbase + ((v >> 2) << 4) + (half << 3) + ((v & 3) << 1)];
        }
#pragma unroll
        for (int tm = 0; tm < 2; ++tm) {
            Frag16 afrag;
            const int mbase = (wr * 32 + tm * 16 + lm) * LDS_STRIDE;
#pragma unroll
            for (int v = 0; v < 8; ++v) { // A 16x32: k = (v/4)*16 + half*8 + (v%4)*2
                const int kb = ((v >> 2) << 4) + (half << 3) + ((v & 3) << 1);
                afrag.u[v] = *(const unsigned int*)&As[cur][mbase + kb];
            }
#pragma unroll
            for (int tn = 0; tn < 4; ++tn)
                acc[tm][tn] = __builtin_amdgcn_wmma_f32_16x16x32_bf16(
                    false, afrag.v, false, bfrag[tn].v,
                    (short)0, acc[tm][tn], false, false);
        }

        // commit register-staged data for the next tile, drain async copies,
        // one barrier per K-step
        if (hasNext) { commitA(nxt); commitB(nxt); }
        if (ANY_ASYNC) wait_async_zero();
        __syncthreads();
    }

    // ---- epilogue: C/D layout m = r + 8*half, n = lane&15 ----
    unsigned short* Cb = (unsigned short*)Cg_ + zo * cOuter + zi * cInner;
    float*          Cf = (float*)Cg_          + zo * cOuter + zi * cInner;
#pragma unroll
    for (int tm = 0; tm < 2; ++tm) {
#pragma unroll
        for (int tn = 0; tn < 4; ++tn) {
            const int col  = n0 + wc * 64 + tn * 16 + lm;
            const float bv = bias ? bias[col] : 0.0f;
#pragma unroll
            for (int r = 0; r < 8; ++r) {
                const int row = m0 + wr * 32 + tm * 16 + half * 8 + r;
                const float val = acc[tm][tn][r] * alpha + bv;
                if (OUT_BF16) {
                    Cb[(long long)row * ldc + col] = f32_to_bf16_bits(val);
                } else {
                    // fp32 outputs (scores logits / final out) are streaming:
                    // don't pollute L2's useful working set
                    __builtin_nontemporal_store(val, &Cf[(long long)row * ldc + col]);
                }
            }
        }
    }
}

__global__ void __launch_bounds__(256)
f32_to_bf16_kernel(const float* __restrict__ src, unsigned short* __restrict__ dst,
                   long long n)
{
    long long i = (long long)blockIdx.x * 256 + threadIdx.x;
    const long long stride = (long long)gridDim.x * 256;
    for (; i < n; i += stride) dst[i] = f32_to_bf16_bits(src[i]);
}

// In-place softmax over rows of length 2048; one 256-thread block per row.
// Whole row held in registers: exactly one global read + one write per element.
__global__ void __launch_bounds__(256)
softmax_rows2048_kernel(float* __restrict__ scores)
{
    constexpr int ROW = 2048, PER = 8;
    __shared__ float red[256];
    const long long base = (long long)blockIdx.x * ROW;
    const int t = threadIdx.x;

    float v[PER];
#pragma unroll
    for (int i = 0; i < PER; ++i) v[i] = scores[base + t + i * 256];

    float m = v[0];
#pragma unroll
    for (int i = 1; i < PER; ++i) m = fmaxf(m, v[i]);
    red[t] = m; __syncthreads();
    for (int s = 128; s > 0; s >>= 1) {
        if (t < s) red[t] = fmaxf(red[t], red[t + s]);
        __syncthreads();
    }
    m = red[0]; __syncthreads();

    float sum = 0.0f;
#pragma unroll
    for (int i = 0; i < PER; ++i) { v[i] = __expf(v[i] - m); sum += v[i]; }
    red[t] = sum; __syncthreads();
    for (int s = 128; s > 0; s >>= 1) {
        if (t < s) red[t] += red[t + s];
        __syncthreads();
    }
    const float inv = 1.0f / red[0];
#pragma unroll
    for (int i = 0; i < PER; ++i) scores[base + t + i * 256] = v[i] * inv;
}

extern "C" void kernel_launch(void* const* d_in, const int* in_sizes, int n_in,
                              void* d_out, int out_size, void* d_ws, size_t ws_size,
                              hipStream_t stream)
{
    (void)in_sizes; (void)n_in; (void)out_size; (void)ws_size;
    constexpr int  BATCH = 4, S = 2048, D = 2048, H = 16, HD = 128;
    constexpr long long BS = (long long)BATCH * S;       // 8192 rows
    const float SCALE = 0.08838834764831845f;            // 1/sqrt(128)

    const float* x  = (const float*)d_in[0];
    const float* Wq = (const float*)d_in[1];
    const float* bq = (const float*)d_in[2];
    const float* Wk = (const float*)d_in[3];
    const float* bk = (const float*)d_in[4];
    const float* Wv = (const float*)d_in[5];
    const float* bv = (const float*)d_in[6];
    const float* Wo = (const float*)d_in[7];
    const float* bo = (const float*)d_in[8];

    float* out    = (float*)d_out;
    float* scores = out + BS * D;   // [B,H,S,S] region of the tuple output

    // workspace carve-up (element counts all multiples of 8 -> 16B alignment)
    unsigned short* p   = (unsigned short*)d_ws;
    unsigned short* xb  = p; p += BS * D;
    unsigned short* Wqb = p; p += (long long)D * D;
    unsigned short* Wkb = p; p += (long long)D * HD;
    unsigned short* Wvb = p; p += (long long)D * HD;
    unsigned short* Wob = p; p += (long long)D * D;
    unsigned short* Qb  = p; p += BS * D;
    unsigned short* Kb  = p; p += BS * HD;
    unsigned short* Vb  = p; p += BS * HD;
    unsigned short* AOb = p; p += BS * D;

    auto cvt = [&](const float* s, unsigned short* d, long long n) {
        long long nb = (n + 255) / 256; if (nb > 8192) nb = 8192;
        f32_to_bf16_kernel<<<dim3((unsigned)nb), dim3(256), 0, stream>>>(s, d, n);
    };
    cvt(x,  xb,  BS * D);
    cvt(Wq, Wqb, (long long)D * D);
    cvt(Wk, Wkb, (long long)D * HD);
    cvt(Wv, Wvb, (long long)D * HD);
    cvt(Wo, Wob, (long long)D * D);

    const dim3 blk(256);

    // Q = x @ Wq + bq  -> bf16 [BS, D]
    gemm_wmma_kernel<false, false, true>
        <<<dim3(D / 128, BS / 128, 1), blk, 0, stream>>>(
        xb, Wqb, bq, Qb, D, D, D, D, D, 1.0f,
        1, 0LL, 0LL, 0LL, 0LL, 0LL, 0LL);

    // K = x @ Wk + bk  -> bf16 [BS, HD] (single shared head)
    gemm_wmma_kernel<false, false, true>
        <<<dim3(HD / 128, BS / 128, 1), blk, 0, stream>>>(
        xb, Wkb, bk, Kb, HD, D, D, HD, HD, 1.0f,
        1, 0LL, 0LL, 0LL, 0LL, 0LL, 0LL);

    // V = x @ Wv + bv  -> bf16 [BS, HD]
    gemm_wmma_kernel<false, false, true>
        <<<dim3(HD / 128, BS / 128, 1), blk, 0, stream>>>(
        xb, Wvb, bv, Vb, HD, D, D, HD, HD, 1.0f,
        1, 0LL, 0LL, 0LL, 0LL, 0LL, 0LL);

    // scores = SCALE * Q_h @ K^T per (b,h); K shared across heads (bInner=0)
    gemm_wmma_kernel<true, false, false>
        <<<dim3(S / 128, S / 128, BATCH * H), blk, 0, stream>>>(
        Qb, Kb, nullptr, scores, S, HD, D, HD, S, SCALE,
        H, (long long)S * D, (long long)HD,
           (long long)S * HD, 0LL,
           (long long)H * S * S, (long long)S * S);

    // in-place softmax over last axis: B*H*S rows of length S
    softmax_rows2048_kernel<<<dim3(BATCH * H * S), blk, 0, stream>>>(scores);

    // attn = P @ V per (b,h): fp32 A path (reads post-softmax scores directly)
    gemm_wmma_kernel<false, true, true>
        <<<dim3(HD / 128, S / 128, BATCH * H), blk, 0, stream>>>(
        scores, Vb, nullptr, AOb, HD, S, S, HD, D, 1.0f,
        H, (long long)H * S * S, (long long)S * S,
           (long long)S * HD, 0LL,
           (long long)S * D, (long long)HD);

    // out = attn @ Wo + bo -> fp32 [BS, D]
    gemm_wmma_kernel<false, false, false>
        <<<dim3(D / 128, BS / 128, 1), blk, 0, stream>>>(
        AOb, Wob, bo, out, D, D, D, D, D, 1.0f,
        1, 0LL, 0LL, 0LL, 0LL, 0LL, 0LL);
}
